// TemporalSelfAttention_10617159156446
// MI455X (gfx1250) — matrix operand order repaired and checked by reading, not compiled, not run on hardware
//
#include <hip/hip_runtime.h>
#include <hip/hip_bf16.h>

#define NQ     22500
#define BEVH   150
#define BEVW   150
#define EMBED  256
#define HEADS  8
#define HDIM   32
#define QUEUE  2
#define POINTS 4

typedef __attribute__((ext_vector_type(16))) __bf16 v16bf;
typedef __attribute__((ext_vector_type(8)))  __bf16 v8bf;
typedef __attribute__((ext_vector_type(8)))  float  v8f;

__device__ __forceinline__ __bf16 f2bf(float f) {
  union { float f; unsigned u; } x; x.f = f;
  // round-to-nearest-even fp32 -> bf16
  unsigned r = x.u + 0x7FFFu + ((x.u >> 16) & 1u);
  unsigned short h = (unsigned short)(r >> 16);
  __bf16 b;
  __builtin_memcpy(&b, &h, sizeof(b));
  return b;
}

// Generic GEMM: C[M,N] = A[M,K] * B[K,N] + bias[N] (+ residual[M,N])
// A(i,k) = (k < K0) ? A0[i*K0+k] : A1[i*(K-K0)+(k-K0)] + A2[i*(K-K0)+(k-K0)]
// Workgroup = 256 threads (8 waves), covers 128 rows x all N columns.
// Full A-tile (128 x K) and B^T (N x K) staged in LDS as bf16 with +8 pad.
__global__ __launch_bounds__(256)
void wmma_gemm_kernel(const float* __restrict__ A0,
                      const float* __restrict__ A1,
                      const float* __restrict__ A2,
                      const float* __restrict__ B,
                      const float* __restrict__ bias,
                      const float* __restrict__ residual,
                      float* __restrict__ C,
                      int M, int N, int K, int K0,
                      int kshift, int nshift)
{
  extern __shared__ __align__(16) char smem_raw[];
  const int lda = K + 8;                       // bf16 elements per padded row
  __bf16* As = (__bf16*)smem_raw;              // [128][lda]
  __bf16* Bs = As + 128 * lda;                 // [N][lda]   (B transposed)

  const int tid = threadIdx.x;
  const int m0  = blockIdx.x * 128;
  const int KE  = K - K0;

  // ---- Stage A tile (coalesced over k) ----
  const int atot = 128 << kshift;              // 128*K
  for (int idx = tid; idx < atot; idx += 256) {
    const int row = idx >> kshift;
    const int k   = idx & (K - 1);
    const int g   = m0 + row;
    float aval = 0.0f;
    if (g < M) {
      if (k < K0) {
        aval = A0[(size_t)g * K0 + k];
      } else {
        const int kk = k - K0;
        aval = A1[(size_t)g * KE + kk] + A2[(size_t)g * KE + kk];
      }
    }
    As[row * lda + k] = f2bf(aval);
  }
  // ---- Stage B transposed (coalesced over n) ----
  const int btot = K << nshift;                // K*N
  for (int idx = tid; idx < btot; idx += 256) {
    const int k = idx >> nshift;
    const int n = idx & (N - 1);
    Bs[n * lda + k] = f2bf(B[idx]);
  }
  __syncthreads();

  const int wave  = tid >> 5;
  const int lane  = tid & 31;
  const int l16   = lane & 15;
  const int lhalf = lane >> 4;                 // 0: lanes 0-15, 1: lanes 16-31
  const __bf16* Arow = As + (wave * 16 + l16) * lda;
  const int mbase = m0 + wave * 16 + lhalf * 8;

  for (int nt = 0; nt < N; nt += 16) {
    const int col = nt + l16;
    const float bn = bias[col];
    v8f acc;
#pragma unroll
    for (int r = 0; r < 8; ++r) acc[r] = bn;
    const __bf16* Bcol = Bs + col * lda;
#pragma unroll 4
    for (int k0 = 0; k0 < K; k0 += 32) {
      // A 16x32 bf16 fragment: lanes 0-15 hold K {0..7,16..23}, lanes 16-31 {8..15,24..31}
      union { v16bf v; v8bf h[2]; } a, b;
      a.h[0] = *(const v8bf*)(Arow + k0 + lhalf * 8);
      a.h[1] = *(const v8bf*)(Arow + k0 + 16 + lhalf * 8);
      // B 32x16 bf16 fragment: lane col = l16; lanes 0-15 rows K {0..7,16..23}, lanes 16-31 {8..15,24..31}
      b.h[0] = *(const v8bf*)(Bcol + k0 + lhalf * 8);
      b.h[1] = *(const v8bf*)(Bcol + k0 + 16 + lhalf * 8);
      acc = __builtin_amdgcn_wmma_f32_16x16x32_bf16(
          false, a.v, false, b.v, (short)0, acc, false, false);
    }
#pragma unroll
    for (int r = 0; r < 8; ++r) {
      const int m = mbase + r;                 // C/D: VGPR r -> row r (+8 for upper lanes)
      if (m < M) {
        float val = acc[r];
        if (residual) val += residual[(size_t)m * N + col];
        C[(size_t)m * N + col] = val;
      }
    }
  }
}

// Deformable sampling: one wave32 per (query, head); lane = channel (HDIM==32).
// Includes the 4-way softmax over points and the mean over QUEUE.
__global__ __launch_bounds__(256)
void msdeform_kernel(const float* __restrict__ v,     // (QUEUE, NQ, EMBED)
                     const float* __restrict__ off,   // (NQ, HEADS, QUEUE, POINTS, 2)
                     const float* __restrict__ aw,    // (NQ, HEADS, QUEUE, POINTS) raw logits
                     const float* __restrict__ ref,   // (QUEUE, NQ, 1, 2)
                     float* __restrict__ out)         // (NQ, EMBED)
{
  const int wid  = blockIdx.x * 8 + (threadIdx.x >> 5);
  const int lane = threadIdx.x & 31;
  if (wid >= NQ * HEADS) return;
  const int h = wid & (HEADS - 1);
  const int i = wid >> 3;

  float acc = 0.0f;
#pragma unroll
  for (int qu = 0; qu < QUEUE; ++qu) {
    // softmax over the 4 points (wave-uniform scalars)
    const float* awp = aw + (size_t)i * (HEADS * QUEUE * POINTS) + (h * QUEUE + qu) * POINTS;
    const float a0 = awp[0], a1 = awp[1], a2 = awp[2], a3 = awp[3];
    const float mx = fmaxf(fmaxf(a0, a1), fmaxf(a2, a3));
    const float e0 = __expf(a0 - mx), e1 = __expf(a1 - mx);
    const float e2 = __expf(a2 - mx), e3 = __expf(a3 - mx);
    const float inv = 1.0f / (e0 + e1 + e2 + e3);
    const float wt[POINTS] = { e0 * inv, e1 * inv, e2 * inv, e3 * inv };

    const float rx = ref[((size_t)qu * NQ + i) * 2 + 0] * (float)BEVW;
    const float ry = ref[((size_t)qu * NQ + i) * 2 + 1] * (float)BEVH;
    const float* offp = off + (size_t)i * (HEADS * QUEUE * POINTS * 2)
                            + (h * QUEUE + qu) * (POINTS * 2);
    const float* vb = v + (size_t)qu * NQ * EMBED + h * HDIM + lane;

#pragma unroll
    for (int p = 0; p < POINTS; ++p) {
      // norm == (W,H) == sampling scale, so loc math collapses:
      const float x = rx + offp[p * 2 + 0] - 0.5f;
      const float y = ry + offp[p * 2 + 1] - 0.5f;
      const float xf = floorf(x), yf = floorf(y);
      const float fx = x - xf, fy = y - yf;
      const int ix = (int)xf, iy = (int)yf;
      const bool vx0 = (ix >= 0) && (ix < BEVW);
      const bool vx1 = (ix + 1 >= 0) && (ix + 1 < BEVW);
      const bool vy0 = (iy >= 0) && (iy < BEVH);
      const bool vy1 = (iy + 1 >= 0) && (iy + 1 < BEVH);
      float s00 = 0.f, s01 = 0.f, s10 = 0.f, s11 = 0.f;
      if (vy0 && vx0) s00 = vb[(size_t)(iy * BEVW + ix) * EMBED];
      if (vy0 && vx1) s01 = vb[(size_t)(iy * BEVW + ix + 1) * EMBED];
      if (vy1 && vx0) s10 = vb[(size_t)((iy + 1) * BEVW + ix) * EMBED];
      if (vy1 && vx1) s11 = vb[(size_t)((iy + 1) * BEVW + ix + 1) * EMBED];
      const float samp = s00 * (1.f - fx) * (1.f - fy) + s01 * fx * (1.f - fy)
                       + s10 * (1.f - fx) * fy        + s11 * fx * fy;
      acc = fmaf(wt[p], samp, acc);
    }
  }
  out[(size_t)i * EMBED + h * HDIM + lane] = acc * (1.0f / QUEUE);
}

extern "C" void kernel_launch(void* const* d_in, const int* in_sizes, int n_in,
                              void* d_out, int out_size, void* d_ws, size_t ws_size,
                              hipStream_t stream) {
  (void)in_sizes; (void)n_in; (void)out_size; (void)ws_size;
  const float* query     = (const float*)d_in[0];
  const float* query_pos = (const float*)d_in[1];
  const float* value     = (const float*)d_in[2];
  const float* ref       = (const float*)d_in[3];
  // d_in[4] spatial_shapes, d_in[5] level_start_index: fixed (150,150)/(0) constants
  const float* W_off     = (const float*)d_in[6];
  const float* b_off     = (const float*)d_in[7];
  const float* W_attn    = (const float*)d_in[8];
  const float* b_attn    = (const float*)d_in[9];
  const float* W_val     = (const float*)d_in[10];
  const float* b_val     = (const float*)d_in[11];
  const float* W_out     = (const float*)d_in[12];
  const float* b_out     = (const float*)d_in[13];
  float* out = (float*)d_out;

  float* v_ws    = (float*)d_ws;                                     // 45000*256
  float* off_ws  = v_ws   + (size_t)QUEUE * NQ * EMBED;              // 22500*128
  float* aw_ws   = off_ws + (size_t)NQ * HEADS * QUEUE * POINTS * 2; // 22500*64
  float* samp_ws = aw_ws  + (size_t)NQ * HEADS * QUEUE * POINTS;     // 22500*256

  const dim3 blk(256);
  auto lds_bytes = [](int N, int K) -> size_t {
    return ((size_t)128 * (size_t)(K + 8) + (size_t)N * (size_t)(K + 8)) * 2u;
  };

  // 1) v = value @ W_val + b_val          (M=45000, N=256, K=256)
  {
    const int M = QUEUE * NQ, N = EMBED, K = EMBED;
    wmma_gemm_kernel<<<dim3((M + 127) / 128), blk, lds_bytes(N, K), stream>>>(
        value, nullptr, nullptr, W_val, b_val, nullptr, v_ws,
        M, N, K, K, 8, 8);
  }
  // 2) off = [value0, q+qpos] @ W_off + b_off   (M=22500, N=128, K=512, K0=256)
  {
    const int M = NQ, N = 128, K = 2 * EMBED, K0 = EMBED;
    wmma_gemm_kernel<<<dim3((M + 127) / 128), blk, lds_bytes(N, K), stream>>>(
        value, query, query_pos, W_off, b_off, nullptr, off_ws,
        M, N, K, K0, 9, 7);
  }
  // 3) aw_raw = [value0, q+qpos] @ W_attn + b_attn  (N=64)
  {
    const int M = NQ, N = 64, K = 2 * EMBED, K0 = EMBED;
    wmma_gemm_kernel<<<dim3((M + 127) / 128), blk, lds_bytes(N, K), stream>>>(
        value, query, query_pos, W_attn, b_attn, nullptr, aw_ws,
        M, N, K, K0, 9, 6);
  }
  // 4) deformable sampling + softmax + queue mean  (22500 blocks x 8 waves)
  msdeform_kernel<<<dim3(NQ), blk, 0, stream>>>(v_ws, off_ws, aw_ws, ref, samp_ws);
  // 5) out = samp @ W_out + b_out + query  (residual epilogue)
  {
    const int M = NQ, N = EMBED, K = EMBED;
    wmma_gemm_kernel<<<dim3((M + 127) / 128), blk, lds_bytes(N, K), stream>>>(
        samp_ws, nullptr, nullptr, W_out, b_out, query, out,
        M, N, K, K, 8, 8);
  }
}
